// Text2graph_77695958385206
// MI455X (gfx1250) — matrix operand order repaired and checked by reading, not compiled, not run on hardware
//
#include <hip/hip_runtime.h>
#include <hip/hip_bf16.h>

typedef __attribute__((ext_vector_type(16))) __bf16 v16bf;
typedef __attribute__((ext_vector_type(4)))  __bf16 v4bf;
typedef __attribute__((ext_vector_type(8)))  float  v8f;

#define WMMA_BF16(a, b, c) \
    __builtin_amdgcn_wmma_f32_16x16x32_bf16(false, (a), false, (b), (short)0, (c), false, false)

// ---------------------------------------------------------------------------
// Problem constants (from reference)
// ---------------------------------------------------------------------------
#define BATCH 2
#define HEADS 16
#define DMODEL 1024
#define SEQ 2048              // LQ == LK
#define HDIM 64               // DK == DV
#define MROWS (BATCH * SEQ)   // 4096

__device__ __forceinline__ v8f v8f_zero() {
    v8f z;
#pragma unroll
    for (int j = 0; j < 8; j++) z[j] = 0.0f;
    return z;
}

__device__ __forceinline__ v4bf pack4(float4 v) {
    v4bf p;
    p[0] = (__bf16)v.x; p[1] = (__bf16)v.y;
    p[2] = (__bf16)v.z; p[3] = (__bf16)v.w;
    return p;
}

__device__ __forceinline__ void put4(v16bf& f, int base, float4 v) {
    f[base + 0] = (__bf16)v.x; f[base + 1] = (__bf16)v.y;
    f[base + 2] = (__bf16)v.z; f[base + 3] = (__bf16)v.w;
}

// ---------------------------------------------------------------------------
// GEMM: C[M,N] = A[M,K] @ W[N,K]^T + bias[N] (+ res[M,N])
// 64x64 tile, BK=64, 128 threads = 4 wave32. bf16 WMMA, f32 accumulate.
// float4 global loads -> packed bf16x4 LDS stores; prefetch of next K-chunk.
// ---------------------------------------------------------------------------
__global__ __launch_bounds__(128)
void mha_gemm_nt_bias(const float* __restrict__ A, const float* __restrict__ W,
                      const float* __restrict__ bias, const float* __restrict__ res,
                      float* __restrict__ C, int M, int N, int K)
{
    __shared__ __bf16 sA[64 * 64];
    __shared__ __bf16 sB[64 * 64];

    const int t    = threadIdx.x;
    const int wave = t >> 5;
    const int lane = t & 31;
    const int ln16 = lane & 15;
    const int hi   = lane >> 4;
    const int m0   = blockIdx.x * 64;
    const int n0   = blockIdx.y * 64;

    v8f acc[4];
#pragma unroll
    for (int r = 0; r < 4; r++) acc[r] = v8f_zero();

    for (int kc = 0; kc < K; kc += 64) {
        // prefetch next K-chunk while this one is consumed (global_prefetch_b8)
        if (kc + 64 < K) {
            int pr = t >> 1, pc = (t & 1) * 32;
            __builtin_prefetch(A + (size_t)(m0 + pr) * K + kc + 64 + pc, 0, 1);
            __builtin_prefetch(W + (size_t)(n0 + pr) * K + kc + 64 + pc, 0, 1);
        }
        // cooperative staging: float4 global loads, bf16x4 packed LDS stores
#pragma unroll
        for (int i = 0; i < 8; i++) {
            int idx4 = i * 128 + t;             // 1024 float4 = 64x64 tile
            int r  = idx4 >> 4;                 // 16 float4 per row
            int c4 = (idx4 & 15) << 2;
            float4 va = *(const float4*)(A + (size_t)(m0 + r) * K + kc + c4);
            float4 vb = *(const float4*)(W + (size_t)(n0 + r) * K + kc + c4);
            *(v4bf*)&sA[idx4 << 2] = pack4(va);
            *(v4bf*)&sB[idx4 << 2] = pack4(vb);
        }
        __syncthreads();

#pragma unroll
        for (int kk = 0; kk < 64; kk += 32) {
            v16bf af;
            const int arow = wave * 16 + ln16;
#pragma unroll
            for (int i = 0; i < 16; i++) {
                int k = kk + ((i >> 3) << 4) + hi * 8 + (i & 7);
                af[i] = sA[arow * 64 + k];
            }
#pragma unroll
            for (int nt = 0; nt < 4; nt++) {
                v16bf bf;
                const int bn = nt * 16 + ln16;
#pragma unroll
                for (int i = 0; i < 16; i++) {
                    int k = kk + ((i >> 3) << 4) + hi * 8 + (i & 7);
                    bf[i] = sB[bn * 64 + k];
                }
                acc[nt] = WMMA_BF16(af, bf, acc[nt]);
            }
        }
        __syncthreads();
    }

    // epilogue: bias + optional residual, f32 store
#pragma unroll
    for (int nt = 0; nt < 4; nt++) {
        int n = n0 + nt * 16 + ln16;
        float bv = bias ? bias[n] : 0.0f;
#pragma unroll
        for (int j = 0; j < 8; j++) {
            int m = m0 + wave * 16 + hi * 8 + j;
            float v = acc[nt][j] + bv;
            if (res) v += res[(size_t)m * N + n];
            C[(size_t)m * N + n] = v;
        }
    }
}

// ---------------------------------------------------------------------------
// A-fragment (16x32 bf16, rows = q) from a row pointer: per lane two
// contiguous float8 runs -> four float4 loads.
// ---------------------------------------------------------------------------
__device__ __forceinline__ v16bf load_afrag(const float* __restrict__ rowbase,
                                            int half, int hi)
{
    const float4* r4 = (const float4*)rowbase;
    int b0 = half * 8 + (hi << 1);        // d = half*32 + hi*8
    float4 a0 = r4[b0], a1 = r4[b0 + 1];
    float4 a2 = r4[b0 + 4], a3 = r4[b0 + 5];   // d + 16
    v16bf f;
    put4(f, 0, a0); put4(f, 4, a1); put4(f, 8, a2); put4(f, 12, a3);
    return f;
}

// B-fragment (32x16 bf16, from K^T): per lane two contiguous float8 runs.
__device__ __forceinline__ v16bf load_bfrag_kt(const float* __restrict__ rowbase,
                                               int half, int hi)
{
    const float4* r4 = (const float4*)rowbase;
    int b0 = half * 8 + (hi << 1);        // d = half*32 + hi*8, then d + 16
    v16bf f;
    put4(f, 0,  r4[b0 + 0]);
    put4(f, 4,  r4[b0 + 1]);
    put4(f, 8,  r4[b0 + 4]);
    put4(f, 12, r4[b0 + 5]);
    return f;
}

// ---------------------------------------------------------------------------
// Pass 1: per-(b,h,k-column) online max / sum-of-exp over the QUERY axis.
// One wave handles 16 K-columns; streams 128 q-tiles of 16 rows.
// ---------------------------------------------------------------------------
__global__ __launch_bounds__(32)
void mha_colstats(const float* __restrict__ qp, const float* __restrict__ kp,
                  const int* __restrict__ mask,
                  float* __restrict__ colmax, float* __restrict__ colsum)
{
    const int bid  = blockIdx.x;          // B * H * (SEQ/16) = 4096
    const int kt   = bid & 127;
    const int h    = (bid >> 7) & 15;
    const int b    = bid >> 11;
    const int lane = threadIdx.x;
    const int ln16 = lane & 15;
    const int hi   = lane >> 4;
    const float scale = 0.125f;           // 1/sqrt(64)
    const float NEGINF = -__builtin_inff();

    const int kcol = kt * 16 + ln16;
    const float* kb = kp + ((size_t)(b * SEQ + kcol)) * DMODEL + h * HDIM;

    // B = K^T fragments, resident for the whole q loop
    v16bf bk[2];
    bk[0] = load_bfrag_kt(kb, 0, hi);
    bk[1] = load_bfrag_kt(kb, 1, hi);

    float m = NEGINF, s = 0.0f;

    for (int qt = 0; qt < 128; qt++) {
        const int qrow = qt * 16 + ln16;
        const float* qb = qp + ((size_t)(b * SEQ + qrow)) * DMODEL + h * HDIM;
        v8f c = v8f_zero();
        c = WMMA_BF16(load_afrag(qb, 0, hi), bk[0], c);
        c = WMMA_BF16(load_afrag(qb, 1, hi), bk[1], c);

        // this lane holds 8 rows of ONE column; online column softmax stats
        const int4* m4 = (const int4*)(mask + b * SEQ + qt * 16 + hi * 8);
        int4 ma = m4[0], mb4 = m4[1];
        int mr[8] = {ma.x, ma.y, ma.z, ma.w, mb4.x, mb4.y, mb4.z, mb4.w};

        float vals[8];
        float tm = NEGINF;
#pragma unroll
        for (int j = 0; j < 8; j++) {
            float v = mr[j] ? c[j] * scale : NEGINF;
            vals[j] = v;
            tm = fmaxf(tm, v);
        }
        float mn = fmaxf(m, tm);
        if (mn > NEGINF) {
            float acc = 0.0f;
#pragma unroll
            for (int j = 0; j < 8; j++) acc += __expf(vals[j] - mn);
            s = s * __expf(m - mn) + acc;
            m = mn;
        }
    }

    // merge the two 8-row halves (lanes L and L^16 share a column)
    float mo = __shfl_xor(m, 16, 32);
    float so = __shfl_xor(s, 16, 32);
    float mn = fmaxf(m, mo);
    float st = 0.0f;
    if (mn > NEGINF)
        st = s * __expf(m - mn) + so * __expf(mo - mn);

    if (hi == 0) {
        size_t idx = ((size_t)(b * HEADS + h)) * SEQ + kcol;
        colmax[idx] = mn;
        colsum[idx] = st;
    }
}

// ---------------------------------------------------------------------------
// Pass 2: O[b,q,h,:] = sum_k P[q,k] * V[b,k,h,:], P = exp(S-m_k)/s_k (col-norm),
// masked query rows produce zero P. One wave per 16 query rows.
// S recomputed by WMMA; P re-shaped C->A layout through LDS; V tile staged
// in LDS with coalesced row loads, fragment-gathered transposed.
// ---------------------------------------------------------------------------
__global__ __launch_bounds__(32)
void mha_pv(const float* __restrict__ qp, const float* __restrict__ kp,
            const float* __restrict__ vp, const int* __restrict__ mask,
            const float* __restrict__ colmax, const float* __restrict__ colsum,
            float* __restrict__ ao)
{
    __shared__ __bf16 sp[16 * 32];        // P tile (q x k)
    __shared__ __bf16 sV[32 * 64];        // V tile (k x d)

    const int bid  = blockIdx.x;          // B * H * (SEQ/16) = 4096
    const int qt   = bid & 127;
    const int h    = (bid >> 7) & 15;
    const int b    = bid >> 11;
    const int lane = threadIdx.x;
    const int ln16 = lane & 15;
    const int hi   = lane >> 4;
    const float scale = 0.125f;

    // Q fragments for this wave's 16 rows, resident for the whole k loop
    const int qrow = qt * 16 + ln16;
    const float* qb = qp + ((size_t)(b * SEQ + qrow)) * DMODEL + h * HDIM;
    v16bf aq[2];
    aq[0] = load_afrag(qb, 0, hi);
    aq[1] = load_afrag(qb, 1, hi);

    const int4* m4 = (const int4*)(mask + b * SEQ + qt * 16 + hi * 8);
    int4 ma = m4[0], mb4 = m4[1];
    int mrow[8] = {ma.x, ma.y, ma.z, ma.w, mb4.x, mb4.y, mb4.z, mb4.w};

    v8f o[4];
#pragma unroll
    for (int nt = 0; nt < 4; nt++) o[nt] = v8f_zero();

    for (int kt = 0; kt < 64; kt++) {     // 32 k-columns per step
        // --- S subtiles -> normalized P -> LDS (C layout -> A layout) ---
#pragma unroll
        for (int sub = 0; sub < 2; sub++) {
            const int kc = kt * 32 + sub * 16 + ln16;
            const float* kb = kp + ((size_t)(b * SEQ + kc)) * DMODEL + h * HDIM;
            v8f c = v8f_zero();
            c = WMMA_BF16(aq[0], load_bfrag_kt(kb, 0, hi), c);
            c = WMMA_BF16(aq[1], load_bfrag_kt(kb, 1, hi), c);

            const size_t sidx = ((size_t)(b * HEADS + h)) * SEQ + kc;
            const float mcol = colmax[sidx];
            const float rs   = 1.0f / colsum[sidx];
#pragma unroll
            for (int j = 0; j < 8; j++) {
                float p = 0.0f;
                if (mrow[j]) p = __expf(c[j] * scale - mcol) * rs;
                sp[(hi * 8 + j) * 32 + sub * 16 + ln16] = (__bf16)p;
            }
        }
        // --- stage V tile: one coalesced 64-float row per lane ---
        {
            const int kidx = kt * 32 + lane;
            const float4* vrow = (const float4*)(vp + ((size_t)(b * SEQ + kidx)) * DMODEL + h * HDIM);
#pragma unroll
            for (int c4 = 0; c4 < 16; c4++)
                *(v4bf*)&sV[lane * 64 + c4 * 4] = pack4(vrow[c4]);
        }
        __syncthreads();

        // reload P as a 16x32 bf16 A fragment
        v16bf ap;
#pragma unroll
        for (int i = 0; i < 16; i++) {
            int kk = ((i >> 3) << 4) + hi * 8 + (i & 7);
            ap[i] = sp[ln16 * 32 + kk];
        }
        // V B-fragments from LDS (transposed gather)
#pragma unroll
        for (int nt = 0; nt < 4; nt++) {
            v16bf bv;
#pragma unroll
            for (int i = 0; i < 16; i++)
                bv[i] = sV[(((i >> 3) << 4) + hi * 8 + (i & 7)) * 64 + nt * 16 + ln16];
            o[nt] = WMMA_BF16(ap, bv, o[nt]);
        }
        __syncthreads();
    }

    // store attention output [4096, 1024] at column block h*64
#pragma unroll
    for (int nt = 0; nt < 4; nt++)
#pragma unroll
        for (int j = 0; j < 8; j++) {
            int q = qt * 16 + hi * 8 + j;
            ao[((size_t)(b * SEQ + q)) * DMODEL + h * HDIM + nt * 16 + ln16] = o[nt][j];
        }
}

// ---------------------------------------------------------------------------
// Launch
// ---------------------------------------------------------------------------
extern "C" void kernel_launch(void* const* d_in, const int* in_sizes, int n_in,
                              void* d_out, int out_size, void* d_ws, size_t ws_size,
                              hipStream_t stream)
{
    const float* Q    = (const float*)d_in[0];
    const float* Kin  = (const float*)d_in[1];
    const float* Vin  = (const float*)d_in[2];
    const int*   mask = (const int*)  d_in[3];
    const float* W_Q  = (const float*)d_in[4];
    const float* b_Q  = (const float*)d_in[5];
    const float* W_K  = (const float*)d_in[6];
    const float* b_K  = (const float*)d_in[7];
    const float* W_V  = (const float*)d_in[8];
    const float* b_V  = (const float*)d_in[9];
    const float* W_O  = (const float*)d_in[10];
    const float* b_O  = (const float*)d_in[11];
    float* out = (float*)d_out;

    const size_t PROJ = (size_t)MROWS * DMODEL;   // 4M floats per buffer
    float* ws   = (float*)d_ws;
    float* qp   = ws;
    float* kp   = ws + PROJ;
    float* vp   = ws + 2 * PROJ;
    float* ao   = ws + 3 * PROJ;
    float* cmax = ws + 4 * PROJ;
    float* csum = cmax + (size_t)BATCH * HEADS * SEQ;

    dim3 gg(MROWS / 64, DMODEL / 64);   // 64 x 16
    dim3 gb(128);

    // input projections (f32 in, bf16 WMMA, f32 out)
    mha_gemm_nt_bias<<<gg, gb, 0, stream>>>(Q,   W_Q, b_Q, nullptr, qp, MROWS, DMODEL, DMODEL);
    mha_gemm_nt_bias<<<gg, gb, 0, stream>>>(Kin, W_K, b_K, nullptr, kp, MROWS, DMODEL, DMODEL);
    mha_gemm_nt_bias<<<gg, gb, 0, stream>>>(Vin, W_V, b_V, nullptr, vp, MROWS, DMODEL, DMODEL);

    // column-axis softmax stats (over query dim), then P@V
    mha_colstats<<<BATCH * HEADS * (SEQ / 16), 32, 0, stream>>>(qp, kp, mask, cmax, csum);
    mha_pv<<<BATCH * HEADS * (SEQ / 16), 32, 0, stream>>>(qp, kp, vp, mask, cmax, csum, ao);

    // output projection + bias + residual, straight into d_out
    mha_gemm_nt_bias<<<gg, gb, 0, stream>>>(ao, W_O, b_O, Q, out, MROWS, DMODEL, DMODEL);
}
